// GNNNodeClassifier_38920993636494
// MI455X (gfx1250) — matrix operand order repaired and checked by reading, not compiled, not run on hardware
//
#include <hip/hip_runtime.h>
#include <hip/hip_bf16.h>

// ---------------------------------------------------------------------------
// GCN node classifier for MI455X (gfx1250, wave32).
//   layer: agg = D^-1/2 (A+I) D^-1/2 (X W) + b
//   GEMMs via v_wmma_f32_16x16x32_bf16 (bf16 in, fp32 accumulate).
//   Gather/scatter aggregation via float4 loads + global_atomic_add_f32
//   (h fits in 192MB L2, so gather is L2-BW bound, not HBM bound).
// ---------------------------------------------------------------------------

#define NN 10000
#define EE 640000
#define FF 128
#define HH 128
#define CC 40

typedef __attribute__((ext_vector_type(16))) __bf16 v16bf;
typedef __attribute__((ext_vector_type(8)))  __bf16 v8bf;
typedef __attribute__((ext_vector_type(8)))  float  v8f;

// ---- degree / normalization ------------------------------------------------

__global__ __launch_bounds__(256) void k_init_deg(float* deg) {
    int n = blockIdx.x * blockDim.x + threadIdx.x;
    if (n < NN) deg[n] = 1.0f;  // self-loop contributes 1
}

__global__ __launch_bounds__(256) void k_deg_edges(const int* __restrict__ dst,
                                                   float* __restrict__ deg) {
    int e = blockIdx.x * blockDim.x + threadIdx.x;
    if (e < EE) atomicAdd(&deg[dst[e]], 1.0f);
}

__global__ __launch_bounds__(256) void k_rsqrt_inplace(float* deg) {
    int n = blockIdx.x * blockDim.x + threadIdx.x;
    if (n < NN) deg[n] = rsqrtf(deg[n]);
}

// ---- precision conversion / weight transpose -------------------------------

__global__ __launch_bounds__(256) void k_f32_to_bf16(const float* __restrict__ in,
                                                     __bf16* __restrict__ out,
                                                     int count) {
    int i = blockIdx.x * blockDim.x + threadIdx.x;
    if (i < count) out[i] = (__bf16)in[i];
}

__global__ __launch_bounds__(256) void k_relu_f32_to_bf16(const float* __restrict__ in,
                                                          __bf16* __restrict__ out,
                                                          int count) {
    int i = blockIdx.x * blockDim.x + threadIdx.x;
    if (i < count) out[i] = (__bf16)fmaxf(in[i], 0.0f);
}

// Wt[n*K + k] = W[k*Ncol + n]  (column-major bf16 copy of W)
__global__ __launch_bounds__(256) void k_w_bf16_transpose(const float* __restrict__ W,
                                                          __bf16* __restrict__ Wt,
                                                          int K, int Ncol) {
    int t = blockIdx.x * blockDim.x + threadIdx.x;
    if (t >= K * Ncol) return;
    int k = t / Ncol, n = t % Ncol;
    Wt[(size_t)n * K + k] = (__bf16)W[t];
}

// ---- bf16 WMMA GEMM: Out[M x Ncols] = A[M x K] * B (B given as Bt[Ncols x K])
// One wave per 16x16 output tile; K looped in steps of 32.
__global__ __launch_bounds__(256) void k_gemm_bf16_wmma(const __bf16* __restrict__ A,
                                                        const __bf16* __restrict__ Bt,
                                                        float* __restrict__ Out,
                                                        int M, int Ncols, int K) {
    const int lane = threadIdx.x & 31;
    const int wave = threadIdx.x >> 5;
    const int colTiles = Ncols >> 4;
    const int tile = blockIdx.x * (blockDim.x >> 5) + wave;
    const int rowT = tile / colTiles;
    const int colT = tile % colTiles;
    if (rowT * 16 >= M) return;  // wave-uniform: EXEC stays all-ones

    const int m    = lane & 15;  // A-row / B-col / D-col index within tile
    const int half = lane >> 4;

    const __bf16* Arow = A  + (size_t)(rowT * 16 + m) * K;
    const __bf16* Bcol = Bt + (size_t)(colT * 16 + m) * K;

    v8f acc = {};
#pragma unroll
    for (int k0 = 0; k0 < HH; k0 += 32) {
        // A fragment (contiguous K order, matching B): K = k0+16*half+[0..15]
        v16bf a = *(const v16bf*)(Arow + k0 + 16 * half);
        // B fragment: lanes 0-15 hold K = k0+[0..15], lanes 16-31 K = k0+16+[0..15]
        // for column n = lane&15; contiguous in column-major Bt.
        v16bf b = *(const v16bf*)(Bcol + k0 + 16 * half);

        acc = __builtin_amdgcn_wmma_f32_16x16x32_bf16(
            /*neg_a=*/false, a, /*neg_b=*/false, b,
            /*c_mod=*/(short)0, acc, /*reuse_a=*/false, /*reuse_b=*/false);
    }

    // D layout: lane 0-15 -> N=lane, M=r; lane 16-31 -> N=lane-16, M=8+r
    float* Orow = Out + (size_t)(rowT * 16 + 8 * half) * Ncols + colT * 16 + m;
#pragma unroll
    for (int r = 0; r < 8; ++r)
        Orow[(size_t)r * Ncols] = acc[r];
}

// ---- aggregation -----------------------------------------------------------

// agg[n][f] = h[n][f] * dinv[n]^2 + b[f]   (self-loop term + bias)
__global__ __launch_bounds__(256) void k_agg_init(const float* __restrict__ h,
                                                  const float* __restrict__ dinv,
                                                  const float* __restrict__ b,
                                                  float* __restrict__ agg) {
    int i = blockIdx.x * blockDim.x + threadIdx.x;
    if (i >= NN * HH) return;
    int n = i >> 7, f = i & 127;
    float di = dinv[n];
    agg[i] = h[i] * di * di + b[f];
}

// one wave per edge; each lane handles 4 consecutive features (32*4 = 128)
__global__ __launch_bounds__(256) void k_agg_edges(const float* __restrict__ h,
                                                   const int* __restrict__ src,
                                                   const int* __restrict__ dst,
                                                   const float* __restrict__ dinv,
                                                   float* __restrict__ agg) {
    long long t = (long long)blockIdx.x * blockDim.x + threadIdx.x;
    int e = (int)(t >> 5);
    int lane = (int)(t & 31);
    if (e >= EE) return;
    int s = src[e], d = dst[e];
    float w = dinv[s] * dinv[d];
    float4 v = ((const float4*)(h + (size_t)s * HH))[lane];  // L2-resident gather
    float* o = agg + (size_t)d * HH + lane * 4;
    atomicAdd(o + 0, v.x * w);
    atomicAdd(o + 1, v.y * w);
    atomicAdd(o + 2, v.z * w);
    atomicAdd(o + 3, v.w * w);
}

// ---- classifier + log_softmax ----------------------------------------------

__global__ __launch_bounds__(256) void k_classifier(const float* __restrict__ h,
                                                    const float* __restrict__ Wc,
                                                    const float* __restrict__ bc,
                                                    float* __restrict__ out) {
    int t = blockIdx.x * blockDim.x + threadIdx.x;
    if (t >= NN * CC) return;
    int n = t / CC, c = t % CC;
    const float* hr = h + (size_t)n * HH;
    float s = bc[c];
#pragma unroll 8
    for (int k = 0; k < HH; ++k)
        s = fmaf(hr[k], Wc[(size_t)k * CC + c], s);
    out[t] = s;
}

__global__ __launch_bounds__(256) void k_log_softmax(float* __restrict__ out) {
    int n = blockIdx.x * blockDim.x + threadIdx.x;
    if (n >= NN) return;
    float* row = out + (size_t)n * CC;
    float m = row[0];
#pragma unroll
    for (int c = 1; c < CC; ++c) m = fmaxf(m, row[c]);
    float s = 0.0f;
#pragma unroll
    for (int c = 0; c < CC; ++c) s += expf(row[c] - m);
    float l = logf(s);
#pragma unroll
    for (int c = 0; c < CC; ++c) row[c] = row[c] - m - l;
}

// ---------------------------------------------------------------------------

extern "C" void kernel_launch(void* const* d_in, const int* in_sizes, int n_in,
                              void* d_out, int out_size, void* d_ws, size_t ws_size,
                              hipStream_t stream) {
    const float* x  = (const float*)d_in[0];
    const int*   ei = (const int*)d_in[1];   // [2, E] int32
    const float* W1 = (const float*)d_in[2];
    const float* b1 = (const float*)d_in[3];
    const float* W2 = (const float*)d_in[4];
    const float* b2 = (const float*)d_in[5];
    const float* Wc = (const float*)d_in[6];
    const float* bc = (const float*)d_in[7];
    float* out = (float*)d_out;

    const int* src = ei;        // edge_index[0]
    const int* dst = ei + EE;   // edge_index[1]

    // workspace bump allocator (256B aligned)
    char* w = (char*)d_ws;
    size_t o = 0;
    auto bump = [&](size_t bytes) {
        size_t r = o;
        o = (o + bytes + 255) & ~(size_t)255;
        return r;
    };
    float*   dinv = (float*)(w + bump((size_t)NN * 4));
    __bf16*  xb   = (__bf16*)(w + bump((size_t)NN * FF * 2));
    __bf16*  w1t  = (__bf16*)(w + bump((size_t)FF * HH * 2));
    __bf16*  w2t  = (__bf16*)(w + bump((size_t)HH * HH * 2));
    float*   h1   = (float*)(w + bump((size_t)NN * HH * 4));
    float*   agg1 = (float*)(w + bump((size_t)NN * HH * 4));
    __bf16*  hb   = (__bf16*)(w + bump((size_t)NN * HH * 2));
    float*   h2   = (float*)(w + bump((size_t)NN * HH * 4));
    float*   agg2 = (float*)(w + bump((size_t)NN * HH * 4));
    (void)ws_size; (void)in_sizes; (void)n_in; (void)out_size;

    const int B = 256;
    auto gsz = [](long long n, int b) { return (unsigned)((n + b - 1) / b); };

    // ---- normalization prepass
    k_init_deg<<<gsz(NN, B), B, 0, stream>>>(dinv);
    k_deg_edges<<<gsz(EE, B), B, 0, stream>>>(dst, dinv);
    k_rsqrt_inplace<<<gsz(NN, B), B, 0, stream>>>(dinv);

    // ---- precision conversion
    k_f32_to_bf16<<<gsz((long long)NN * FF, B), B, 0, stream>>>(x, xb, NN * FF);
    k_w_bf16_transpose<<<gsz((long long)FF * HH, B), B, 0, stream>>>(W1, w1t, FF, HH);
    k_w_bf16_transpose<<<gsz((long long)HH * HH, B), B, 0, stream>>>(W2, w2t, HH, HH);

    // ---- layer 1: h1 = x @ W1 (WMMA), agg1 = norm-aggregate(h1) + b1
    {
        long long tiles = (long long)(NN / 16) * (HH / 16);  // 625 * 8
        k_gemm_bf16_wmma<<<gsz(tiles, 8), B, 0, stream>>>(xb, w1t, h1, NN, HH, FF);
    }
    k_agg_init<<<gsz((long long)NN * HH, B), B, 0, stream>>>(h1, dinv, b1, agg1);
    k_agg_edges<<<gsz((long long)EE * 32, B), B, 0, stream>>>(h1, src, dst, dinv, agg1);

    // ---- relu + convert for layer 2
    k_relu_f32_to_bf16<<<gsz((long long)NN * HH, B), B, 0, stream>>>(agg1, hb, NN * HH);

    // ---- layer 2
    {
        long long tiles = (long long)(NN / 16) * (HH / 16);
        k_gemm_bf16_wmma<<<gsz(tiles, 8), B, 0, stream>>>(hb, w2t, h2, NN, HH, HH);
    }
    k_agg_init<<<gsz((long long)NN * HH, B), B, 0, stream>>>(h2, dinv, b2, agg2);
    k_agg_edges<<<gsz((long long)EE * 32, B), B, 0, stream>>>(h2, src, dst, dinv, agg2);

    // ---- classifier + log_softmax (fp32 for softmax accuracy)
    k_classifier<<<gsz((long long)NN * CC, B), B, 0, stream>>>(agg2, Wc, bc, out);
    k_log_softmax<<<gsz(NN, B), B, 0, stream>>>(out);
}
